// HardGateMOE_57466662420622
// MI455X (gfx1250) — compile-verified, matrix operand order checked by reading, not run on hardware
//
#include <hip/hip_runtime.h>
#include <hip/hip_bf16.h>
#include <math.h>

// Problem constants (from reference)
#define NDIM 2048
#define HDIM 1024
#define FDIM 4096
#define EDIM 8
#define KSEL 2
#define MTILE 32
#define MAXP (NDIM * KSEL)   // worst-case pairs per expert
#define HS_STRIDE 264        // h LDS row stride (bf16 elems), padded

typedef __attribute__((ext_vector_type(8)))  float  v8f;
typedef __attribute__((ext_vector_type(8)))  __bf16 v8bf;
typedef __attribute__((ext_vector_type(16))) __bf16 v16bf;

union Frag16 { v16bf v; v8bf h[2]; };

// Blocked bf16 weight layout: [E][N/16 ntile][K/32 ktile][lane 0..31][16 elems]
// chunk for lane l: elems 0..7  = (n = n0+(l&15), k = k0+(l>>4)*8 + i)
//                   elems 8..15 = (n = n0+(l&15), k = k0+(l>>4)*8 + 16 + i)
// -> a WMMA B fragment is two contiguous b128 loads per lane.
#define TILE_ELEMS 512          // 32 lanes * 16 bf16
#define W1_NT_STRIDE ((HDIM / 32) * TILE_ELEMS)   // elems between n-tiles of W1
#define W2_NT_STRIDE ((FDIM / 32) * TILE_ELEMS)   // elems between n-tiles of W2

// Branch-free tanh-approx gelu: tanh(u) = sign(u)*(1-e)/(1+e), e = exp(-2|u|)
__device__ __forceinline__ float gelu_tanh(float v) {
    const float c = 0.7978845608028654f; // sqrt(2/pi)
    float u = c * v * (1.0f + 0.044715f * v * v);
    float e = __expf(-2.0f * __builtin_fabsf(u));
    float th = (1.0f - e) * __builtin_amdgcn_rcpf(1.0f + e);
    th = __builtin_copysignf(th, u);
    return 0.5f * v * (1.0f + th);
}

// A fragment (16x32 bf16, M x K) from LDS, row-major with given stride.
__device__ __forceinline__ v16bf load_a_lds(const __bf16* base, int stride,
                                            int mrow, int k0, int lane) {
    int r  = mrow + (lane & 15);
    int kk = k0 + ((lane >> 4) << 3);
    const __bf16* p = base + r * stride + kk;
    Frag16 f;
    f.h[0] = *(const v8bf*)(p);
    f.h[1] = *(const v8bf*)(p + 16);
    return f.v;
}

// B fragment from pre-blocked bf16 weights: two contiguous b128 loads.
__device__ __forceinline__ v16bf load_b_blocked(const __bf16* __restrict__ p) {
    Frag16 f;
    f.h[0] = *(const v8bf*)(p);
    f.h[1] = *(const v8bf*)(p + 8);
    return f.v;
}

__device__ __forceinline__ v8f wmma_bf16(v16bf a, v16bf b, v8f c) {
    return __builtin_amdgcn_wmma_f32_16x16x32_bf16(
        false, a, false, b, (short)0, c, false, false);
}

// ---------------------------------------------------------------- kernel 0
__global__ void zero_kernel(float* __restrict__ out, int* __restrict__ counts) {
    size_t i = (size_t)blockIdx.x * blockDim.x + threadIdx.x;
    if (i < (size_t)NDIM * HDIM) out[i] = 0.0f;
    if (i < EDIM) counts[i] = 0;
}

// ---------------------------------------------------------------- pre-pass
// x (f32 row-major) -> xbf (bf16 row-major), 8 elems/thread
__global__ void cvt_x_kernel(const float* __restrict__ x,
                             __bf16* __restrict__ xbf) {
    size_t i = ((size_t)blockIdx.x * 256 + threadIdx.x) * 8;
    float4 a = *(const float4*)(x + i);
    float4 b = *(const float4*)(x + i + 4);
    v8bf pk;
    pk[0] = (__bf16)a.x; pk[1] = (__bf16)a.y; pk[2] = (__bf16)a.z; pk[3] = (__bf16)a.w;
    pk[4] = (__bf16)b.x; pk[5] = (__bf16)b.y; pk[6] = (__bf16)b.z; pk[7] = (__bf16)b.w;
    *(v8bf*)(xbf + i) = pk;
}

// W (f32, [E][K][N] row-major) -> blocked bf16 fragment layout.
// One thread produces one 16-elem lane-chunk (32B contiguous store).
__global__ void cvt_w_kernel(const float* __restrict__ W,
                             __bf16* __restrict__ Wb, int Kd, int Nd) {
    size_t idx = (size_t)blockIdx.x * 256 + threadIdx.x;
    int lane = (int)(idx & 31);
    size_t t = idx >> 5;
    int nkt = Kd >> 5;
    int nnt = Nd >> 4;
    int ktile = (int)(t % nkt); t /= nkt;
    int ntile = (int)(t % nnt); t /= nnt;
    int e = (int)t;
    const float* src = W + (size_t)e * Kd * Nd;
    int n  = ntile * 16 + (lane & 15);
    int kk = ktile * 32 + ((lane >> 4) << 3);
    __bf16* dst = Wb + ((size_t)e * nnt * nkt + (size_t)ntile * nkt + ktile) * TILE_ELEMS
                     + lane * 16;
    v8bf p0, p1;
#pragma unroll
    for (int i = 0; i < 8; ++i) p0[i] = (__bf16)src[(size_t)(kk + i) * Nd + n];
#pragma unroll
    for (int i = 0; i < 8; ++i) p1[i] = (__bf16)src[(size_t)(kk + 16 + i) * Nd + n];
    *(v8bf*)(dst)     = p0;
    *(v8bf*)(dst + 8) = p1;
}

// ---------------------------------------------------------------- kernel 1
__global__ void gate_logits_kernel(const float* __restrict__ x,
                                   const float* __restrict__ Wg,
                                   float* __restrict__ logits) {
    int wave = threadIdx.x >> 5;
    int lane = threadIdx.x & 31;
    int n = blockIdx.x * 8 + wave;
    if (n >= NDIM) return;
    float acc[EDIM];
#pragma unroll
    for (int e = 0; e < EDIM; ++e) acc[e] = 0.0f;
    for (int h = lane; h < HDIM; h += 32) {
        float xv = x[(size_t)n * HDIM + h];
#pragma unroll
        for (int e = 0; e < EDIM; ++e) acc[e] += xv * Wg[(size_t)e * HDIM + h];
    }
#pragma unroll
    for (int e = 0; e < EDIM; ++e) {
        for (int off = 16; off > 0; off >>= 1)
            acc[e] += __shfl_xor(acc[e], off, 32);
    }
    if (lane == 0) {
#pragma unroll
        for (int e = 0; e < EDIM; ++e) logits[(size_t)n * EDIM + e] = acc[e];
    }
}

// ---------------------------------------------------------------- kernel 2
__global__ void gate_stats_kernel(const float* __restrict__ logits,
                                  float* __restrict__ colmax,
                                  float* __restrict__ colsum) {
    __shared__ float red[256];
    int e = blockIdx.x;
    int tid = threadIdx.x;
    float m = -1e30f;
    for (int n = tid; n < NDIM; n += 256)
        m = fmaxf(m, logits[(size_t)n * EDIM + e]);
    red[tid] = m; __syncthreads();
    for (int s = 128; s > 0; s >>= 1) {
        if (tid < s) red[tid] = fmaxf(red[tid], red[tid + s]);
        __syncthreads();
    }
    float mm = red[0]; __syncthreads();
    float sacc = 0.0f;
    for (int n = tid; n < NDIM; n += 256)
        sacc += expf(logits[(size_t)n * EDIM + e] - mm);
    red[tid] = sacc; __syncthreads();
    for (int s = 128; s > 0; s >>= 1) {
        if (tid < s) red[tid] += red[tid + s];
        __syncthreads();
    }
    if (tid == 0) { colmax[e] = mm; colsum[e] = red[0]; }
}

// ---------------------------------------------------------------- kernel 3
__global__ void route_kernel(const int* __restrict__ mapping,
                             const float* __restrict__ logits,
                             const float* __restrict__ colmax,
                             const float* __restrict__ colsum,
                             int* __restrict__ counts,
                             int* __restrict__ pairTok,
                             float* __restrict__ pairW) {
    int n = blockIdx.x * 256 + threadIdx.x;
    if (n >= NDIM) return;
    int e0 = mapping[n * KSEL + 0];
    int e1 = mapping[n * KSEL + 1];
    float w0 = expf(logits[(size_t)n * EDIM + e0] - colmax[e0]) / colsum[e0];
    float w1 = expf(logits[(size_t)n * EDIM + e1] - colmax[e1]) / colsum[e1];
    float s = w0 + w1;
    w0 /= s; w1 /= s;
    int p0 = atomicAdd(&counts[e0], 1);
    pairTok[e0 * MAXP + p0] = n;
    pairW [e0 * MAXP + p0] = w0;
    int p1 = atomicAdd(&counts[e1], 1);
    pairTok[e1 * MAXP + p1] = n;
    pairW [e1 * MAXP + p1] = w1;
}

// ---------------------------------------------------------------- kernel 4
// Fused routed MoE MLP. Block = 32 pairs of one expert, 8 waves.
// Weights are pre-blocked bf16 (L2-resident); x pre-converted bf16.
__global__ __launch_bounds__(256)
void moe_mlp_kernel(const __bf16* __restrict__ xbf,
                    const __bf16* __restrict__ W1b,
                    const float* __restrict__ b1,
                    const __bf16* __restrict__ W2b,
                    const float* __restrict__ b2,
                    const int* __restrict__ counts,
                    const int* __restrict__ pairTok,
                    const float* __restrict__ pairW,
                    float* __restrict__ out) {
    const int e = blockIdx.y;
    const int cnt = counts[e];
    const int start = blockIdx.x * MTILE;
    if (start >= cnt) return;
    const int mval = min(MTILE, cnt - start);

    extern __shared__ char smem[];
    __bf16* xs = (__bf16*)smem;                                  // 32 x 1024 bf16 (LDS offset 0)
    __bf16* hs = (__bf16*)(smem + MTILE * HDIM * 2);             // 32 x HS_STRIDE bf16
    int*    ts = (int*)  (smem + MTILE * HDIM * 2 + MTILE * HS_STRIDE * 2);
    float*  tw = (float*)(ts + MTILE);

    const int tid  = threadIdx.x;
    const int lane = tid & 31;
    const int wave = tid >> 5;

    if (tid < MTILE) {
        int  t = (tid < mval) ? pairTok[e * MAXP + start + tid] : 0;
        float w = (tid < mval) ? pairW [e * MAXP + start + tid] : 0.0f;
        ts[tid] = t; tw[tid] = w;
    }
    __syncthreads();

    // ---- Async global->LDS staging of the x tile (bf16, 2KB per row).
    // Invalid rows source token 0; their outputs are discarded in the epilogue.
    for (int r = wave; r < MTILE; r += 8) {
        const __bf16* src = xbf + (size_t)ts[r] * HDIM;
        unsigned rowoff = (unsigned)(r * HDIM * 2);  // LDS byte offset (xs at 0)
#pragma unroll
        for (int c = 0; c < 4; ++c) {
            unsigned dst = rowoff + (unsigned)((c * 32 + lane) * 16);
            const __bf16* ga = src + (c * 32 + lane) * 8;
            asm volatile("global_load_async_to_lds_b128 %0, %1, off"
                         :: "v"(dst), "v"(ga) : "memory");
        }
    }
    asm volatile("s_wait_asynccnt 0" ::: "memory");

    const __bf16* W1e = W1b + (size_t)e * (FDIM / 16) * (HDIM / 32) * TILE_ELEMS;
    const __bf16* W2e = W2b + (size_t)e * (HDIM / 16) * (FDIM / 32) * TILE_ELEMS;
    const float* b1e = b1 + (size_t)e * FDIM;
    const float* b2e = b2 + (size_t)e * HDIM;

    v8f acc[16];   // [mt*8 + nt], out tile 32 x 128 per wave
#pragma unroll
    for (int i = 0; i < 16; ++i) acc[i] = (v8f){};

    __syncthreads();

    for (int f0 = 0; f0 < FDIM; f0 += 256) {
        const int fw = f0 + wave * 32;    // this wave's 32-wide f-slice

        // Hoisted bias loads (overlap with GEMM1 WMMA stream)
        float bb0 = b1e[fw + (lane & 15)];
        float bb1 = b1e[fw + 16 + (lane & 15)];

        // ---- GEMM1: h[32 x 32] = x[32 x 1024] @ W1[:, fw:fw+32] (full-H in wave)
        v8f hacc[4];
#pragma unroll
        for (int i = 0; i < 4; ++i) hacc[i] = (v8f){};

        const __bf16* w1p = W1e + (size_t)(fw >> 4) * W1_NT_STRIDE + lane * 16;
        for (int k0 = 0; k0 < HDIM; k0 += 32) {
            const int kt = k0 >> 5;
            __builtin_prefetch(w1p + (size_t)(kt + 4) * TILE_ELEMS, 0, 1);
            v16bf a0  = load_a_lds(xs, HDIM, 0,  k0, lane);
            v16bf a1  = load_a_lds(xs, HDIM, 16, k0, lane);
            v16bf bw0 = load_b_blocked(w1p + (size_t)kt * TILE_ELEMS);
            v16bf bw1 = load_b_blocked(w1p + W1_NT_STRIDE + (size_t)kt * TILE_ELEMS);
            hacc[0] = wmma_bf16(a0, bw0, hacc[0]);
            hacc[1] = wmma_bf16(a0, bw1, hacc[1]);
            hacc[2] = wmma_bf16(a1, bw0, hacc[2]);
            hacc[3] = wmma_bf16(a1, bw1, hacc[3]);
        }

        __syncthreads();   // prior-iteration reads of hs complete before overwrite

        // gelu(+b1) and store bf16 h tile into LDS (branch-free gelu)
#pragma unroll
        for (int mt = 0; mt < 2; ++mt) {
#pragma unroll
            for (int nt = 0; nt < 2; ++nt) {
                v8f c = hacc[mt * 2 + nt];
                int col = wave * 32 + nt * 16 + (lane & 15);
                float bb = nt ? bb1 : bb0;
                int rbase = mt * 16 + ((lane >> 4) << 3);
#pragma unroll
                for (int r = 0; r < 8; ++r) {
                    float v = gelu_tanh(c[r] + bb);
                    hs[(rbase + r) * HS_STRIDE + col] = (__bf16)v;
                }
            }
        }
        __syncthreads();

        // ---- GEMM2: acc[32 x 128] += h[32 x 256] @ W2[f0:f0+256, wave cols]
        const __bf16* w2p = W2e + (size_t)(wave * 8) * W2_NT_STRIDE
                                + (size_t)(f0 >> 5) * TILE_ELEMS + lane * 16;
        for (int k0 = 0; k0 < 256; k0 += 32) {
            const int kt = k0 >> 5;
            __builtin_prefetch(w2p + (size_t)(kt + 8) * TILE_ELEMS, 0, 1);
            v16bf a0 = load_a_lds(hs, HS_STRIDE, 0,  k0, lane);
            v16bf a1 = load_a_lds(hs, HS_STRIDE, 16, k0, lane);
#pragma unroll
            for (int nt = 0; nt < 8; ++nt) {
                v16bf bw = load_b_blocked(w2p + (size_t)nt * W2_NT_STRIDE
                                              + (size_t)kt * TILE_ELEMS);
                acc[nt]     = wmma_bf16(a0, bw, acc[nt]);
                acc[8 + nt] = wmma_bf16(a1, bw, acc[8 + nt]);
            }
        }
    }

    // ---- Epilogue: weighted scatter-add with b2
#pragma unroll
    for (int mt = 0; mt < 2; ++mt) {
        int rbase = mt * 16 + ((lane >> 4) << 3);
#pragma unroll
        for (int nt = 0; nt < 8; ++nt) {
            int col = wave * 128 + nt * 16 + (lane & 15);
            float b2v = b2e[col];
            v8f c = acc[mt * 8 + nt];
#pragma unroll
            for (int r = 0; r < 8; ++r) {
                int row = rbase + r;
                if (row < mval) {
                    atomicAdd(&out[(size_t)ts[row] * HDIM + col],
                              tw[row] * (c[r] + b2v));
                }
            }
        }
    }
}

// ------------------------------------------------------------------ launch
extern "C" void kernel_launch(void* const* d_in, const int* in_sizes, int n_in,
                              void* d_out, int out_size, void* d_ws, size_t ws_size,
                              hipStream_t stream) {
    const float* x       = (const float*)d_in[0];
    const int*   mapping = (const int*)  d_in[1];
    const float* Wg      = (const float*)d_in[2];
    const float* W1      = (const float*)d_in[3];
    const float* b1      = (const float*)d_in[4];
    const float* W2      = (const float*)d_in[5];
    const float* b2      = (const float*)d_in[6];
    float* out = (float*)d_out;

    // -------- workspace layout (byte offsets, 32B aligned regions) --------
    char* wsb = (char*)d_ws;
    float* logits  = (float*)(wsb + 0);            // 65536 B
    float* colmax  = (float*)(wsb + 65536);        // 32 B
    float* colsum  = (float*)(wsb + 65568);        // 32 B
    int*   counts  = (int*)  (wsb + 65600);        // 32 B
    int*   pairTok = (int*)  (wsb + 65664);        // 131072 B
    float* pairW   = (float*)(wsb + 196736);       // 131072 B
    __bf16* xbf    = (__bf16*)(wsb + 327936);      // N*H*2      = 4 MB
    __bf16* W1b    = (__bf16*)(wsb + 327936 + (size_t)NDIM * HDIM * 2);
    __bf16* W2b    = (__bf16*)((char*)W1b + (size_t)EDIM * HDIM * FDIM * 2);

    zero_kernel<<<((size_t)NDIM * HDIM + 255) / 256, 256, 0, stream>>>(out, counts);

    // Pre-pass: bf16 conversions (x row-major; weights into blocked frag layout)
    cvt_x_kernel<<<(NDIM * HDIM) / (256 * 8), 256, 0, stream>>>(x, xbf);
    {
        size_t chunks1 = (size_t)EDIM * (FDIM / 16) * (HDIM / 32) * 32;
        cvt_w_kernel<<<chunks1 / 256, 256, 0, stream>>>(W1, W1b, HDIM, FDIM);
        size_t chunks2 = (size_t)EDIM * (HDIM / 16) * (FDIM / 32) * 32;
        cvt_w_kernel<<<chunks2 / 256, 256, 0, stream>>>(W2, W2b, FDIM, HDIM);
    }

    gate_logits_kernel<<<NDIM / 8, 256, 0, stream>>>(x, Wg, logits);
    gate_stats_kernel<<<EDIM, 256, 0, stream>>>(logits, colmax, colsum);
    route_kernel<<<(NDIM + 255) / 256, 256, 0, stream>>>(mapping, logits, colmax,
                                                         colsum, counts, pairTok, pairW);

    size_t smem = (size_t)MTILE * HDIM * 2      // xs
                + (size_t)MTILE * HS_STRIDE * 2 // hs
                + MTILE * sizeof(int)           // ts
                + MTILE * sizeof(float);        // tw
    dim3 grid(MAXP / MTILE, EDIM);
    moe_mlp_kernel<<<grid, 256, smem, stream>>>(xbf, W1b, b1, W2b, b2,
                                                counts, pairTok, pairW, out);
}
